// MultiHeadAttention_58763742544693
// MI455X (gfx1250) — matrix operand order repaired and checked by reading, not compiled, not run on hardware
//
#include <hip/hip_runtime.h>

// Problem constants (B, S, D, H) = (2, 2048, 1024, 16), DK = 64
#define BB 2
#define SS 2048
#define DD 1024
#define HH 16
#define DK 64

typedef __bf16 v16bf __attribute__((ext_vector_type(16)));
typedef __bf16 v8bf  __attribute__((ext_vector_type(8)));
typedef float  v8f   __attribute__((ext_vector_type(8)));
typedef float  v4f   __attribute__((ext_vector_type(4)));

__device__ __forceinline__ v8f wmma_bf16(v16bf a, v16bf b, v8f c) {
  // D = A(16x32 bf16) * B(32x16 bf16) + C(16x16 f32)
  return __builtin_amdgcn_wmma_f32_16x16x32_bf16(
      /*neg_a=*/false, a, /*neg_b=*/false, b,
      /*c_mod=*/(short)0, c, /*reuse_a=*/false, /*reuse_b=*/false);
}

// ---------------------------------------------------------------------------
// Kernel 1: fused projection  out = relu(X @ W + b), f32 in -> bf16 out
//   grid = (S/64, D/64, 3*B), block = 128 (4 waves, each 16 rows x 64 cols)
//   mat 0/1 (Q,K): store [b,h,s,dk]  -> K^T / Q A-frag friendly
//   mat 2   (V)  : store [b,h,dk,s]  (transposed) -> P*V B-frag friendly
// ---------------------------------------------------------------------------
__global__ __launch_bounds__(128) void proj_relu_bf16(
    const float* __restrict__ q, const float* __restrict__ k,
    const float* __restrict__ v,
    const float* __restrict__ Wq, const float* __restrict__ bq,
    const float* __restrict__ Wk, const float* __restrict__ bk,
    const float* __restrict__ Wv, const float* __restrict__ bv,
    unsigned short* __restrict__ Qo, unsigned short* __restrict__ Ko,
    unsigned short* __restrict__ Vt)
{
  __shared__ __align__(64) __bf16 Wt[64 * 32];   // W^T tile: [n (64)][k (32)]

  const int tid   = threadIdx.x;
  const int wave  = tid >> 5;
  const int lane  = tid & 31;
  const int ln16  = lane & 15;
  const int half8 = (lane >> 4) << 3;            // 0 or 8

  const int mat = blockIdx.z % 3;                // 0=Q 1=K 2=V
  const int b   = blockIdx.z / 3;
  const float* X  = (mat == 0) ? q  : (mat == 1) ? k  : v;
  const float* W  = (mat == 0) ? Wq : (mat == 1) ? Wk : Wv;
  const float* Bi = (mat == 0) ? bq : (mat == 1) ? bk : bv;

  const int n0 = blockIdx.y * 64;
  const int s0 = blockIdx.x * 64 + wave * 16;
  const float* Xrow = X + (size_t)b * SS * DD + (size_t)(s0 + ln16) * DD;
  const int aoff = half8;

  v8f c[4];
  #pragma unroll
  for (int t = 0; t < 4; t++)
    #pragma unroll
    for (int i = 0; i < 8; i++) c[t][i] = 0.0f;

  for (int k0 = 0; k0 < DD; k0 += 32) {
    __syncthreads();
    #pragma unroll
    for (int i = 0; i < 16; i++) {
      int idx = tid + i * 128;                   // 0..2047
      int kk = idx >> 6;
      int nn = idx & 63;
      Wt[nn * 32 + kk] = (__bf16)W[(size_t)(k0 + kk) * DD + (n0 + nn)];
    }
    __syncthreads();

    // A fragment: row M = ln16; lanes<16: K 0..7 & 16..23; lanes>=16: +8
    v4f x0 = *(const v4f*)(Xrow + k0 + aoff);
    v4f x1 = *(const v4f*)(Xrow + k0 + aoff + 4);
    v4f x2 = *(const v4f*)(Xrow + k0 + aoff + 16);
    v4f x3 = *(const v4f*)(Xrow + k0 + aoff + 20);
    v16bf a;
    #pragma unroll
    for (int j = 0; j < 4; j++) {
      a[j]      = (__bf16)x0[j];
      a[4 + j]  = (__bf16)x1[j];
      a[8 + j]  = (__bf16)x2[j];
      a[12 + j] = (__bf16)x3[j];
    }

    #pragma unroll
    for (int t = 0; t < 4; t++) {
      const v8bf b0 = *(const v8bf*)(&Wt[(t * 16 + ln16) * 32 + half8]);
      const v8bf b1 = *(const v8bf*)(&Wt[(t * 16 + ln16) * 32 + half8 + 16]);
      v16bf bfrag;
      #pragma unroll
      for (int j = 0; j < 8; j++) { bfrag[j] = b0[j]; bfrag[8 + j] = b1[j]; }
      c[t] = wmma_bf16(a, bfrag, c[t]);
    }
  }

  // Epilogue: bias + relu + bf16 convert
  if (mat == 2) {
    __bf16* VtB = (__bf16*)Vt;
    #pragma unroll
    for (int t = 0; t < 4; t++) {
      const int nglob = n0 + t * 16 + ln16;
      const float bias = Bi[nglob];
      const int h  = nglob >> 6;
      const int dk = nglob & 63;
      v8bf pk;
      #pragma unroll
      for (int r = 0; r < 8; r++) {
        float val = c[t][r] + bias;
        val = val > 0.0f ? val : 0.0f;
        pk[r] = (__bf16)val;
      }
      // 8 consecutive s per lane -> one 16B store (V stored transposed)
      *(v8bf*)(VtB + ((size_t)(b * HH + h) * DK + dk) * SS + s0 + half8) = pk;
    }
  } else {
    __bf16* O = (mat == 0) ? (__bf16*)Qo : (__bf16*)Ko;
    #pragma unroll
    for (int t = 0; t < 4; t++) {
      const int nglob = n0 + t * 16 + ln16;
      const float bias = Bi[nglob];
      const int h  = nglob >> 6;
      const int dk = nglob & 63;
      #pragma unroll
      for (int r = 0; r < 8; r++) {
        const int sg = s0 + r + half8;
        float val = c[t][r] + bias;
        val = val > 0.0f ? val : 0.0f;
        O[((size_t)(b * HH + h) * SS + sg) * DK + dk] = (__bf16)val;
      }
    }
  }
}

// ---------------------------------------------------------------------------
// Kernel 2: block-cooperative causal flash attention
//   block = 128 threads (4 waves) = 64 query rows of one (b,h)
//   K/V tiles for each 32-key step staged ONCE per block in LDS (4x less L2)
//   grid = B*H*(S/64) = 1024
// ---------------------------------------------------------------------------
__global__ __launch_bounds__(128) void flash_attn_bf16(
    const unsigned short* __restrict__ Qu, const unsigned short* __restrict__ Ku,
    const unsigned short* __restrict__ Vu, float* __restrict__ Out)
{
  __shared__ __align__(64) __bf16 Kt[32 * 64];     // [key (32)][dk (64)]
  __shared__ __align__(64) __bf16 Vs[64 * 32];     // [dk (64)][key (32)]
  __shared__ __align__(64) __bf16 Pb[4 * 16 * 32]; // per-wave P staging

  const __bf16* Qb = (const __bf16*)Qu;
  const __bf16* Kb = (const __bf16*)Ku;
  const __bf16* Vt = (const __bf16*)Vu;

  const int tid   = threadIdx.x;
  const int wave  = tid >> 5;
  const int lane  = tid & 31;
  const int ln16  = lane & 15;
  const int half8 = (lane >> 4) << 3;              // 0 or 8
  const int bh    = blockIdx.x >> 5;               // 0..31 (= b*H + h)
  const int qblk  = (blockIdx.x & 31) << 6;        // 64-row query block
  const int q0    = qblk + wave * 16;              // this wave's 16 rows
  const float scale2 = 0.125f * 1.44269504088896340736f; // (1/sqrt(64))*log2(e)
  const float NEG_INF = -__builtin_inff();

  // Q A-fragments, resident for the whole kv loop
  const __bf16* Qrow = Qb + ((size_t)bh * SS + (q0 + ln16)) * DK;
  v16bf aq0, aq1;
  {
    v8bf t0 = *(const v8bf*)(Qrow + 2 * half8);
    v8bf t1 = *(const v8bf*)(Qrow + 2 * half8 + 8);
    v8bf t2 = *(const v8bf*)(Qrow + 32 + 2 * half8);
    v8bf t3 = *(const v8bf*)(Qrow + 32 + 2 * half8 + 8);
    #pragma unroll
    for (int j = 0; j < 8; j++) {
      aq0[j] = t0[j]; aq0[8 + j] = t1[j];
      aq1[j] = t2[j]; aq1[8 + j] = t3[j];
    }
  }

  float mrow[8], lrow[8];
  v8f acc[4];
  #pragma unroll
  for (int r = 0; r < 8; r++) { mrow[r] = -3.0e38f; lrow[r] = 0.0f; }
  #pragma unroll
  for (int t = 0; t < 4; t++)
    #pragma unroll
    for (int i = 0; i < 8; i++) acc[t][i] = 0.0f;

  __bf16* Pw = &Pb[wave * 16 * 32];

  // Cooperative staging indices (each thread moves 16 contiguous bf16 = 32B)
  const int kRow = tid >> 2, kCol = (tid & 3) << 4;   // K: 32 rows x 64 dk
  const int vRow = tid >> 1, vCol = (tid & 1) << 4;   // V: 64 dk x 32 keys
  const size_t kBase = (size_t)bh * SS * DK;
  const size_t vBase = (size_t)bh * DK * SS;

  const int kbmax = (qblk + 63) >> 5;                 // block-level causal bound
  for (int kb = 0; kb <= kbmax; kb++) {
    const int kv0 = kb << 5;

    // Stage K (32x64) and V^T (64x32) tiles once per block
    {
      v16bf kk = *(const v16bf*)(Kb + kBase + (size_t)(kv0 + kRow) * DK + kCol);
      *(v16bf*)(&Kt[kRow * 64 + kCol]) = kk;
      v16bf vv = *(const v16bf*)(Vt + vBase + (size_t)vRow * SS + kv0 + vCol);
      *(v16bf*)(&Vs[vRow * 32 + vCol]) = vv;
    }
    __syncthreads();

    if (kv0 <= q0 + 15) {                             // wave-uniform skip
      // K^T B-fragments from LDS (N = key, K = dk)
      v16bf bk00 = *(const v16bf*)(&Kt[ln16 * 64 + (half8 ? 16 : 0)]);
      v16bf bk01 = *(const v16bf*)(&Kt[ln16 * 64 + 32 + (half8 ? 16 : 0)]);
      v16bf bk10 = *(const v16bf*)(&Kt[(16 + ln16) * 64 + (half8 ? 16 : 0)]);
      v16bf bk11 = *(const v16bf*)(&Kt[(16 + ln16) * 64 + 32 + (half8 ? 16 : 0)]);

      v8f s0, s1;
      #pragma unroll
      for (int i = 0; i < 8; i++) { s0[i] = 0.0f; s1[i] = 0.0f; }
      s0 = wmma_bf16(aq0, bk00, s0);
      s0 = wmma_bf16(aq1, bk01, s0);
      s1 = wmma_bf16(aq0, bk10, s1);
      s1 = wmma_bf16(aq1, bk11, s1);

      // Online softmax in exp2 domain; -inf mask keeps fully-masked rows at 0
      #pragma unroll
      for (int r = 0; r < 8; r++) {
        const int qi = q0 + r + half8;
        float sc0 = s0[r] * scale2; if (kv0 + ln16 > qi)      sc0 = NEG_INF;
        float sc1 = s1[r] * scale2; if (kv0 + 16 + ln16 > qi) sc1 = NEG_INF;
        float mx = fmaxf(sc0, sc1);
        #pragma unroll
        for (int off = 1; off < 16; off <<= 1)
          mx = fmaxf(mx, __shfl_xor(mx, off, 32));
        const float mnew = fmaxf(mrow[r], mx);        // stays finite
        const float corr = exp2f(mrow[r] - mnew);
        const float p0 = exp2f(sc0 - mnew);           // exp2(-inf)=0
        const float p1 = exp2f(sc1 - mnew);
        float ps = p0 + p1;
        #pragma unroll
        for (int off = 1; off < 16; off <<= 1)
          ps += __shfl_xor(ps, off, 32);
        lrow[r] = lrow[r] * corr + ps;
        mrow[r] = mnew;
        #pragma unroll
        for (int t = 0; t < 4; t++) acc[t][r] *= corr;
        const int pr = r + half8;
        Pw[pr * 32 + ln16]      = (__bf16)p0;
        Pw[pr * 32 + 16 + ln16] = (__bf16)p1;
      }

      asm volatile("s_wait_dscnt 0" ::: "memory");

      // Reload P as A-fragment (K = 32 keys)
      v16bf ap;
      {
        const __bf16* pr = Pw + ln16 * 32 + 2 * half8;
        v8bf t0 = *(const v8bf*)(pr);
        v8bf t1 = *(const v8bf*)(pr + 8);
        #pragma unroll
        for (int j = 0; j < 8; j++) { ap[j] = t0[j]; ap[8 + j] = t1[j]; }
      }

      // acc += P * V from shared V^T tile
      #pragma unroll
      for (int t = 0; t < 4; t++) {
        v16bf bvf = *(const v16bf*)(&Vs[(t * 16 + ln16) * 32 + (half8 ? 16 : 0)]);
        acc[t] = wmma_bf16(ap, bvf, acc[t]);
      }
    }
    __syncthreads();                                  // protect Kt/Vs reuse
  }

  // Normalize and store f32 output: Out[b][qi][h*64 + dk]
  const int b = bh >> 4;
  const int h = bh & 15;
  #pragma unroll
  for (int t = 0; t < 4; t++) {
    const int dcol = h * 64 + t * 16 + ln16;
    #pragma unroll
    for (int r = 0; r < 8; r++) {
      const int qi = q0 + r + half8;
      Out[((size_t)b * SS + qi) * DD + dcol] = acc[t][r] / lrow[r];
    }
  }
}

// ---------------------------------------------------------------------------
extern "C" void kernel_launch(void* const* d_in, const int* in_sizes, int n_in,
                              void* d_out, int out_size, void* d_ws, size_t ws_size,
                              hipStream_t stream) {
  (void)in_sizes; (void)n_in; (void)out_size; (void)ws_size;
  const float* q  = (const float*)d_in[0];
  const float* k  = (const float*)d_in[1];
  const float* v  = (const float*)d_in[2];
  // d_in[3] is the causal tril mask; applied analytically (ki <= qi)
  const float* Wq = (const float*)d_in[4];
  const float* bq = (const float*)d_in[5];
  const float* Wk = (const float*)d_in[6];
  const float* bk = (const float*)d_in[7];
  const float* Wv = (const float*)d_in[8];
  const float* bv = (const float*)d_in[9];

  const size_t elems = (size_t)BB * HH * SS * DK;   // 4M bf16 each
  unsigned short* Qb = (unsigned short*)d_ws;
  unsigned short* Kb = Qb + elems;
  unsigned short* Vt = Kb + elems;                  // total 24 MB of d_ws

  dim3 g1(SS / 64, DD / 64, 3 * BB);                // (32, 16, 6)
  proj_relu_bf16<<<g1, 128, 0, stream>>>(q, k, v, Wq, bq, Wk, bk, Wv, bv,
                                         Qb, Kb, Vt);

  dim3 g2(BB * HH * (SS / 64));                     // 1024 blocks, 64 q-rows each
  flash_attn_bf16<<<g2, 128, 0, stream>>>(Qb, Kb, Vt, (float*)d_out);
}
